// LlamaModelParallelMultiheadAttention_46995532153369
// MI455X (gfx1250) — compile-verified, matrix operand order checked by reading, not run on hardware
//
#include <hip/hip_runtime.h>

// ---------------------------------------------------------------- types
typedef __attribute__((ext_vector_type(16))) __bf16 v16bf;
typedef __attribute__((ext_vector_type(8)))  __bf16 v8bf;
typedef __attribute__((ext_vector_type(4)))  __bf16 v4bf;
typedef __attribute__((ext_vector_type(8)))  float  v8f;
typedef __attribute__((ext_vector_type(4)))  float  v4f;
typedef __attribute__((ext_vector_type(4)))  int    v4i_;

// ---------------------------------------------------------------- sizes
constexpr int T_  = 2048;
constexpr int B_  = 2;
constexpr int E_  = 2048;
constexpr int H_  = 16;
constexpr int D_  = 128;           // head dim
constexpr int R_  = T_ * B_;       // 4096 rows of (T,B,·) flattened
constexpr int F_  = 3 * E_;        // 6144
constexpr int BH_ = B_ * H_;       // 32
constexpr float QK_SCALE = 0.29730177875068026f;  // 128^(-1/4)

// workspace byte offsets
constexpr size_t OFF_Q16  = 0;                               // 16 MB (reused as attn16)
constexpr size_t OFF_W16  = (size_t)R_ * E_ * 2;             // +24 MB (Wqkv bf16)
constexpr size_t OFF_WO16 = OFF_W16 + (size_t)F_ * E_ * 2;   // +8 MB
constexpr size_t OFF_KVQ  = OFF_WO16 + (size_t)E_ * E_ * 2;  // +48 MB (kvq bf16)
constexpr size_t OFF_QH   = OFF_KVQ + (size_t)R_ * F_ * 2;   // +16 MB
constexpr size_t OFF_KH   = OFF_QH + (size_t)BH_ * T_ * D_ * 2;
constexpr size_t OFF_VT   = OFF_KH + (size_t)BH_ * T_ * D_ * 2;

// ---------------------------------------------------------------- async copy (CDNA5)
#if defined(__has_builtin)
#if __has_builtin(__builtin_amdgcn_global_load_async_to_lds_b128)
#define HAVE_ASYNC_LDS 1
#endif
#endif
#ifndef HAVE_ASYNC_LDS
#define HAVE_ASYNC_LDS 0
#endif

#if HAVE_ASYNC_LDS
typedef __attribute__((address_space(1))) v4i_* gptr128;
typedef __attribute__((address_space(3))) v4i_* lptr128;
#define ASYNC16(g, l)                                               \
  __builtin_amdgcn_global_load_async_to_lds_b128(                   \
      (gptr128)(g), (lptr128)(l), 0, 0)
__device__ __forceinline__ void wait_async0() {
#if __has_builtin(__builtin_amdgcn_s_wait_asynccnt)
  __builtin_amdgcn_s_wait_asynccnt(0);
#else
  asm volatile("s_wait_asynccnt 0" ::: "memory");
#endif
}
#else
#define ASYNC16(g, l) (*(v8bf*)(l) = *(const v8bf*)(g))
__device__ __forceinline__ void wait_async0() {}
#endif

// ---------------------------------------------------------------- helpers
__device__ __forceinline__ v8f wmma_bf16(v16bf a, v16bf b, v8f c) {
  // D = A(16x32) * B(32x16) + C, f32 accumulate
  return __builtin_amdgcn_wmma_f32_16x16x32_bf16(false, a, false, b,
                                                 (short)0, c, false, false);
}

// 16-lane-striped bf16 fragment from a row pointer: lane<16 takes
// K={k..k+7, k+16..k+23}; lane>=16 takes K={k+8..k+15, k+24..k+31}.
__device__ __forceinline__ v16bf load_frag(const __bf16* __restrict__ row,
                                           int k, int lane) {
  const int off = k + ((lane & 16) ? 8 : 0);
  union { v16bf v; v8bf h[2]; } u;
  u.h[0] = *(const v8bf*)(row + off);
  u.h[1] = *(const v8bf*)(row + off + 16);
  return u.v;
}

// same fragment shape but from an LDS row (k folded into row pointer)
__device__ __forceinline__ v16bf lds_frag(const __bf16* rowp, int lane) {
  const int off = (lane & 16) ? 8 : 0;
  union { v16bf v; v8bf h[2]; } u;
  u.h[0] = *(const v8bf*)(rowp + off);
  u.h[1] = *(const v8bf*)(rowp + off + 16);
  return u.v;
}

// ---------------------------------------------------------------- cast fp32 -> bf16 (x4)
__global__ __launch_bounds__(256) void cast_f32_bf16(
    const float* __restrict__ src, __bf16* __restrict__ dst, int n4) {
  int i = blockIdx.x * 256 + threadIdx.x;
  if (i < n4) {
    v4f x = *(const v4f*)(src + 4 * (size_t)i);
    *(v4bf*)(dst + 4 * (size_t)i) = __builtin_convertvector(x, v4bf);
  }
}

// ---------------------------------------------------------------- GEMM: C = A * B^T (+bias)
// A: M x K bf16 row-major; B: N x K bf16 row-major.
// Block: 256 thr = 8 waves -> 128(M) x 128(N); wave: 32(M) x 64(N).
// K-step 32, double-buffered LDS staging with async global->LDS copies.
constexpr int LPAD = 40;   // padded LDS row stride (elements): 32 data + 8 pad

__global__ __launch_bounds__(256) void gemm_bf16_nt(
    const __bf16* __restrict__ A, const __bf16* __restrict__ B,
    const float* __restrict__ bias,
    float* __restrict__ Cf, __bf16* __restrict__ Cb,
    int M, int N, int K) {
  __shared__ __bf16 ldsA[2][128 * LPAD];
  __shared__ __bf16 ldsB[2][128 * LPAD];

  const int tid  = threadIdx.x;
  const int lane = tid & 31;
  const int wave = tid >> 5;        // 0..7
  const int wm   = wave >> 1;       // 0..3  (32-row slice)
  const int wn   = wave & 1;        // 0..1  (64-col slice)
  const int l16  = lane & 15;
  const int half = lane >> 4;

  const int row0 = blockIdx.x * 128;
  const int col0 = blockIdx.y * 128;

  // staging: A and B tiles are 128 rows x 4 16B-chunks = 512 chunks each;
  // each thread copies 2 chunks of A and 2 of B (4 async ops / stage).
  auto stage = [&](int buf, int k) {
#pragma unroll
    for (int c = tid; c < 512; c += 256) {
      const int row = c >> 2;
      const int ch  = (c & 3) * 8;
      ASYNC16(A + (size_t)(row0 + row) * K + k + ch,
              &ldsA[buf][row * LPAD + ch]);
      ASYNC16(B + (size_t)(col0 + row) * K + k + ch,
              &ldsB[buf][row * LPAD + ch]);
    }
  };

  v8f acc0[4] = {};   // rows wm*32 + 0..15
  v8f acc1[4] = {};   // rows wm*32 + 16..31
  const int nk = K / 32;
  stage(0, 0);
  for (int it = 0; it < nk; ++it) {
    const int cur = it & 1;
    wait_async0();          // my async copies done (incl. ones for `cur`)
    __syncthreads();        // everyone's copies visible; prev reads done
    if (it + 1 < nk) stage(cur ^ 1, (it + 1) * 32);
    // batch all fragment loads, then burst 8 WMMAs (B-frags reused twice)
    const v16bf a0 = lds_frag(&ldsA[cur][(wm * 32 + l16) * LPAD], lane);
    const v16bf a1 = lds_frag(&ldsA[cur][(wm * 32 + 16 + l16) * LPAD], lane);
    v16bf bfr[4];
#pragma unroll
    for (int i = 0; i < 4; ++i)
      bfr[i] = lds_frag(&ldsB[cur][(wn * 64 + i * 16 + l16) * LPAD], lane);
#pragma unroll
    for (int i = 0; i < 4; ++i) acc0[i] = wmma_bf16(a0, bfr[i], acc0[i]);
#pragma unroll
    for (int i = 0; i < 4; ++i) acc1[i] = wmma_bf16(a1, bfr[i], acc1[i]);
  }
#pragma unroll
  for (int i = 0; i < 4; ++i) {
#pragma unroll
    for (int j = 0; j < 8; ++j) {
      const int r0 = row0 + wm * 32 + j + 8 * half;  // C layout: M = j+8*(lane>=16)
      const int c  = col0 + wn * 64 + i * 16 + l16;  // N = lane%16
      float v0 = acc0[i][j];
      float v1 = acc1[i][j];
      if (bias) { const float bb = bias[c]; v0 += bb; v1 += bb; }
      if (Cb) {
        Cb[(size_t)r0 * N + c] = (__bf16)v0;
        Cb[(size_t)(r0 + 16) * N + c] = (__bf16)v1;
      } else {
        Cf[(size_t)r0 * N + c] = v0;
        Cf[(size_t)(r0 + 16) * N + c] = v1;
      }
    }
  }
}

// ---------------------------------------------------------------- RoPE + repack
// kvq row r holds [k | v | q] (E each). RoPE position = r % T (reference's
// reshape quirk). Writes qh/kh as [bh][t][d], v transposed as vT[bh][d][t].
__global__ __launch_bounds__(256) void rope_pack(
    const __bf16* __restrict__ kvq,
    const float* __restrict__ fcos, const float* __restrict__ fsin,
    __bf16* __restrict__ qh, __bf16* __restrict__ kh, __bf16* __restrict__ vT) {
  const int idx = blockIdx.x * 256 + threadIdx.x;   // (r, h, i) with i = pair
  const int i = idx & 63;
  const int h = (idx >> 6) & (H_ - 1);
  const int r = idx >> 10;
  if (r >= R_) return;
  const int pos = r & (T_ - 1);
  const int t = r >> 1;        // r = t*B + b, B==2
  const int b = r & 1;
  const int bh = b * H_ + h;

  const float c = fcos[pos * (D_ / 2) + i];
  const float s = fsin[pos * (D_ / 2) + i];

  const size_t base = (size_t)r * F_ + h * D_ + 2 * i;
  const float k0 = (float)kvq[base],            k1 = (float)kvq[base + 1];
  const float v0 = (float)kvq[base + E_],       v1 = (float)kvq[base + E_ + 1];
  const float q0 = (float)kvq[base + 2 * E_],   q1 = (float)kvq[base + 2 * E_ + 1];

  const float qo0 = (q0 * c - q1 * s) * QK_SCALE;
  const float qo1 = (q0 * s + q1 * c) * QK_SCALE;
  const float ko0 = (k0 * c - k1 * s) * QK_SCALE;
  const float ko1 = (k0 * s + k1 * c) * QK_SCALE;

  const size_t qk = ((size_t)bh * T_ + t) * D_ + 2 * i;
  qh[qk] = (__bf16)qo0;  qh[qk + 1] = (__bf16)qo1;
  kh[qk] = (__bf16)ko0;  kh[qk + 1] = (__bf16)ko1;

  const size_t vb = ((size_t)bh * D_ + 2 * i) * T_ + t;
  vT[vb] = (__bf16)v0;  vT[vb + T_] = (__bf16)v1;
}

// ---------------------------------------------------------------- flash attention
// One wave per (q-tile of 16 rows, head). Online softmax over 32-key tiles.
__global__ __launch_bounds__(32) void flash_attn(
    const __bf16* __restrict__ qh, const __bf16* __restrict__ kh,
    const __bf16* __restrict__ vT, __bf16* __restrict__ attn) {
  __shared__ __bf16 pS[16 * 32];   // exp'd probs, staged C-layout -> A-layout

  const int lane = threadIdx.x;
  const int l16  = lane & 15;
  const int half = lane >> 4;
  const int qt = blockIdx.x;       // 0..T/16-1
  const int bh = blockIdx.y;       // 0..31
  const int b  = bh >> 4;          // bh = b*H + h, H = 16
  const int h  = bh & 15;

  // Q fragments (16 rows x 128), A-layout: row = lane%16
  const __bf16* qrow = qh + ((size_t)bh * T_ + qt * 16 + l16) * D_;
  v16bf qf[4];
#pragma unroll
  for (int kk = 0; kk < 4; ++kk) qf[kk] = load_frag(qrow, kk * 32, lane);

  v8f o[8] = {};                   // O: 8 d-tiles of 16x16 (C layout)
  float mrow[8], lrow[8];
#pragma unroll
  for (int j = 0; j < 8; ++j) { mrow[j] = -3.0e38f; lrow[j] = 0.0f; }

  const int qgmax = qt * 16 + 15;
  for (int sb = 0; sb <= qgmax; sb += 32) {
    // --- batch-load all 8 K fragments, then 8 score WMMAs --------------
    const __bf16* krow0 = kh + ((size_t)bh * T_ + sb + l16) * D_;
    if (sb + 32 <= qgmax) __builtin_prefetch(krow0 + (size_t)32 * D_, 0, 1);
    v16bf kf[8];
#pragma unroll
    for (int kk = 0; kk < 4; ++kk) {
      kf[kk]     = load_frag(krow0, kk * 32, lane);
      kf[kk + 4] = load_frag(krow0 + (size_t)16 * D_, kk * 32, lane);
    }
    v8f s0 = {}, s1 = {};
#pragma unroll
    for (int kk = 0; kk < 4; ++kk) s0 = wmma_bf16(qf[kk], kf[kk], s0);
#pragma unroll
    for (int kk = 0; kk < 4; ++kk) s1 = wmma_bf16(qf[kk], kf[kk + 4], s1);

    // --- issue V fragment loads early (independent of softmax) --------
    v16bf vf[8];
#pragma unroll
    for (int dt = 0; dt < 8; ++dt)
      vf[dt] = load_frag(vT + ((size_t)bh * D_ + dt * 16 + l16) * T_ + sb,
                         0, lane);

    // --- causal mask + online softmax, per row j -----------------------
#pragma unroll
    for (int j = 0; j < 8; ++j) {
      const int m  = j + 8 * half;
      const int qg = qt * 16 + m;
      float x0 = s0[j];
      float x1 = s1[j];
      if (sb + l16      > qg) x0 = -3.0e38f;
      if (sb + 16 + l16 > qg) x1 = -3.0e38f;
      float tmax = fmaxf(x0, x1);
#pragma unroll
      for (int d = 1; d < 16; d <<= 1) tmax = fmaxf(tmax, __shfl_xor(tmax, d, 32));
      const float mnew  = fmaxf(mrow[j], tmax);
      const float alpha = __expf(mrow[j] - mnew);
      const float p0 = __expf(x0 - mnew);
      const float p1 = __expf(x1 - mnew);
      float psum = p0 + p1;
#pragma unroll
      for (int d = 1; d < 16; d <<= 1) psum += __shfl_xor(psum, d, 32);
      lrow[j] = lrow[j] * alpha + psum;
      mrow[j] = mnew;
#pragma unroll
      for (int dt = 0; dt < 8; ++dt) o[dt][j] *= alpha;   // rescale row j
      pS[m * 32 + l16]      = (__bf16)p0;
      pS[m * 32 + 16 + l16] = (__bf16)p1;
    }
    __syncthreads();
    // --- P (A-layout 16x32) from LDS -----------------------------------
    union { v16bf v; v8bf hh[2]; } pf;
    pf.hh[0] = *(const v8bf*)(&pS[l16 * 32 + half * 8]);
    pf.hh[1] = *(const v8bf*)(&pS[l16 * 32 + half * 8 + 16]);
    // --- O += P * V -----------------------------------------------------
#pragma unroll
    for (int dt = 0; dt < 8; ++dt) o[dt] = wmma_bf16(pf.v, vf[dt], o[dt]);
    __syncthreads();
  }
  // --- normalize + store to (T, B, E) layout as bf16 -------------------
#pragma unroll
  for (int j = 0; j < 8; ++j) {
    const int m  = j + 8 * half;
    const int tg = qt * 16 + m;
    const float inv = 1.0f / lrow[j];
    const size_t ob = ((size_t)tg * B_ + b) * E_ + h * D_;
#pragma unroll
    for (int dt = 0; dt < 8; ++dt)
      attn[ob + dt * 16 + l16] = (__bf16)(o[dt][j] * inv);
  }
}

// ---------------------------------------------------------------- launcher
extern "C" void kernel_launch(void* const* d_in, const int* in_sizes, int n_in,
                              void* d_out, int out_size, void* d_ws, size_t ws_size,
                              hipStream_t stream) {
  const float* query = (const float*)d_in[0];   // (T,B,E)
  const float* Wqkv  = (const float*)d_in[1];   // (3E,E)
  const float* bqkv  = (const float*)d_in[2];   // (3E)
  const float* Wo    = (const float*)d_in[3];   // (E,E)
  // d_in[4] = bo : unused by the reference output
  const float* fcos  = (const float*)d_in[5];   // (T,D/2)
  const float* fsin  = (const float*)d_in[6];   // (T,D/2)
  float* out = (float*)d_out;                   // (T,B,E) fp32

  char* ws = (char*)d_ws;
  __bf16* q16    = (__bf16*)(ws + OFF_Q16);
  __bf16* w16    = (__bf16*)(ws + OFF_W16);
  __bf16* wo16   = (__bf16*)(ws + OFF_WO16);
  __bf16* kvq16  = (__bf16*)(ws + OFF_KVQ);
  __bf16* qhBuf  = (__bf16*)(ws + OFF_QH);
  __bf16* khBuf  = (__bf16*)(ws + OFF_KH);
  __bf16* vTBuf  = (__bf16*)(ws + OFF_VT);
  __bf16* attn16 = (__bf16*)(ws + OFF_Q16);     // alias: q16 dead after QKV GEMM

  // 1) casts to bf16
  {
    int n4 = R_ * E_ / 4;
    cast_f32_bf16<<<(n4 + 255) / 256, 256, 0, stream>>>(query, q16, n4);
    n4 = F_ * E_ / 4;
    cast_f32_bf16<<<(n4 + 255) / 256, 256, 0, stream>>>(Wqkv, w16, n4);
    n4 = E_ * E_ / 4;
    cast_f32_bf16<<<(n4 + 255) / 256, 256, 0, stream>>>(Wo, wo16, n4);
  }
  // 2) QKV projection: kvq = q16 * Wqkv^T + bqkv   (bf16 out)
  {
    dim3 grid(R_ / 128, F_ / 128);
    gemm_bf16_nt<<<grid, 256, 0, stream>>>(q16, w16, bqkv, nullptr, kvq16,
                                           R_, F_, E_);
  }
  // 3) RoPE + scale + repack (q,k -> [bh][t][d]; v -> [bh][d][t])
  {
    int n = R_ * H_ * (D_ / 2);
    rope_pack<<<(n + 255) / 256, 256, 0, stream>>>(kvq16, fcos, fsin,
                                                   qhBuf, khBuf, vTBuf);
  }
  // 4) causal flash attention -> attn16 in (T,B,E) layout
  {
    dim3 grid(T_ / 16, BH_);
    flash_attn<<<grid, 32, 0, stream>>>(qhBuf, khBuf, vTBuf, attn16);
  }
  // 5) output projection: out = attn16 * Wo^T   (fp32 out, no bias)
  {
    dim3 grid(R_ / 128, E_ / 128);
    gemm_bf16_nt<<<grid, 256, 0, stream>>>(attn16, wo16, nullptr, out, nullptr,
                                           R_, E_, E_);
  }
}
